// image_multihead_attation_block_5042291606104
// MI455X (gfx1250) — compile-verified, matrix operand order checked by reading, not run on hardware
//
#include <hip/hip_runtime.h>
#include <hip/hip_bf16.h>

typedef _Float16 half_t;
typedef half_t v16h __attribute__((ext_vector_type(16)));
typedef half_t v8h  __attribute__((ext_vector_type(8)));
typedef float  v8f  __attribute__((ext_vector_type(8)));

union V16U { v16h v; v8h h[2]; };

__device__ __forceinline__ v16h make_frag(const half_t* p0, const half_t* p1) {
    V16U u;
    u.h[0] = *(const v8h*)p0;
    u.h[1] = *(const v8h*)p1;
    return u.v;
}

__device__ __forceinline__ v8f wmma16(v16h a, v16h b, v8f c) {
    return __builtin_amdgcn_wmma_f32_16x16x32_f16(false, a, false, b, (short)0, c,
                                                  false, false);
}

// ---------------------------------------------------------------------------
// Kernel 1: GroupNorm statistics. One block per (b, g); 16 channels x 1024 px.
// ---------------------------------------------------------------------------
__global__ __launch_bounds__(256) void gn_stats_k(const float* __restrict__ q,
                                                  float* __restrict__ stats) {
    const int bid = blockIdx.x;              // 0..255 : b*32 + g
    const int b = bid >> 5, g = bid & 31;
    const int t = threadIdx.x;
    const size_t base = ((size_t)b * 512 + (size_t)g * 16) * 1024;
    float s = 0.f, ss = 0.f;
    for (int idx = t; idx < 16384; idx += 256) {
        float v = q[base + idx];
        s += v; ss += v * v;
    }
    __shared__ float r1[256], r2[256];
    r1[t] = s; r2[t] = ss;
    __syncthreads();
    for (int st = 128; st > 0; st >>= 1) {
        if (t < st) { r1[t] += r1[t + st]; r2[t] += r2[t + st]; }
        __syncthreads();
    }
    if (t == 0) {
        float mean = r1[0] * (1.f / 16384.f);
        float var  = r2[0] * (1.f / 16384.f) - mean * mean;
        stats[bid * 2]     = mean;
        stats[bid * 2 + 1] = rsqrtf(var + 1e-5f);
    }
}

// ---------------------------------------------------------------------------
// Kernel 2: fold GN affine into projection weights.
//   plainT == 0 : Bt[n][c] = gnw[c]*W[c][n] (fp16), bias[n] = b[n] + sum gnb[c]*W[c][n]
//   plainT == 1 : Bt[n][k] = W[k][n]        (fp16), bias[n] = b[n]
// ---------------------------------------------------------------------------
__global__ __launch_bounds__(256) void fold_w_k(const float* __restrict__ gnw,
                                                const float* __restrict__ gnb,
                                                const float* __restrict__ W,
                                                const float* __restrict__ bvec,
                                                half_t* __restrict__ Bt,
                                                float* __restrict__ biasOut,
                                                int plainT) {
    const int n = blockIdx.x;                // 0..511 output column
    const int t = threadIdx.x;
    __shared__ float red[256];
    float part = 0.f;
    if (!plainT) {
        for (int c = t; c < 512; c += 256) {
            float wv = W[(size_t)c * 512 + n];
            Bt[(size_t)n * 512 + c] = (half_t)(gnw[c] * wv);
            part += gnb[c] * wv;
        }
    } else {
        for (int k = t; k < 512; k += 256)
            Bt[(size_t)n * 512 + k] = (half_t)W[(size_t)k * 512 + n];
    }
    red[t] = part;
    __syncthreads();
    for (int st = 128; st > 0; st >>= 1) {
        if (t < st) red[t] += red[t + st];
        __syncthreads();
    }
    if (t == 0) biasOut[n] = bvec[n] + red[0];
}

// ---------------------------------------------------------------------------
// Kernel 3: normalize + tokenize: Xn[b*1024+s][c] = (q[b][c][s]-mu)*rstd  (fp16)
// ---------------------------------------------------------------------------
__global__ __launch_bounds__(256) void xn_tok_k(const float* __restrict__ q,
                                                const float* __restrict__ stats,
                                                half_t* __restrict__ Xn) {
    const int bs = blockIdx.x * 256 + threadIdx.x;   // 0..8191
    const int b = bs >> 10, s = bs & 1023;
    const float* qb = q + (size_t)b * 512 * 1024 + s;
    half_t* xr = Xn + (size_t)bs * 512;
    for (int c0 = 0; c0 < 512; c0 += 8) {
        const int g = c0 >> 4;
        const float mu   = stats[(b * 32 + g) * 2];
        const float rstd = stats[(b * 32 + g) * 2 + 1];
        v8h hv;
#pragma unroll
        for (int i = 0; i < 8; ++i)
            hv[i] = (half_t)((qb[(size_t)(c0 + i) * 1024] - mu) * rstd);
        *(v8h*)(xr + c0) = hv;
    }
}

// ---------------------------------------------------------------------------
// Kernel 4/6: WMMA GEMM. C[M=8192, N=512] = A[M][512] * Bt[N][512]^T + bias[N].
// Block = 256 thr = 8 waves; block tile 64(M) x 128(N); wave tile 16 x 64.
// Software-pipelined (double-buffered fragments). Bias staged to LDS via
// global_load_async_to_lds_b128 (ASYNCcnt path), waited only at the epilogue.
// mode 0: store fp16 Q/K layout [b][h][s][64]   (value = (acc+bias)*scale)
// mode 1: store fp16 V  layout [b][h][d][1024]
// mode 2: store fp32 out layout [b][c][1024]
// ---------------------------------------------------------------------------
__global__ __launch_bounds__(256) void gemm512_k(const half_t* __restrict__ A,
                                                 const half_t* __restrict__ Bt,
                                                 const float* __restrict__ bias,
                                                 void* __restrict__ Out,
                                                 int mode, float scale) {
    __shared__ float bias_l[512];
    // ---- async-stage the 2KB bias vector into LDS (latency hidden by loop)
    if (threadIdx.x < 128) {
        unsigned ldsoff = (unsigned)(size_t)(&bias_l[0]) + threadIdx.x * 16u;
        const float* gp = bias + threadIdx.x * 4;
        asm volatile("global_load_async_to_lds_b128 %0, %1, off"
                     :: "v"(ldsoff), "v"(gp) : "memory");
    }

    const int lane = threadIdx.x & 31;
    const int w    = threadIdx.x >> 5;
    const int l15  = lane & 15;
    const int off  = (lane & 16) ? 8 : 0;
    const int roff = (lane & 16) ? 8 : 0;
    const int m16  = blockIdx.x * 64 + (w & 3) * 16;
    const int n64  = blockIdx.y * 128 + (w >> 2) * 64;

    const half_t* arow = A  + (size_t)(m16 + l15) * 512;
    const half_t* brow = Bt + (size_t)(n64 + l15) * 512;

    auto ldA = [&](int k) {
        return make_frag(arow + k + off, arow + k + 16 + off);
    };
    auto ldB = [&](int f, int k) {
        const half_t* bp = brow + (size_t)f * 16 * 512 + k;
        return make_frag(bp + off, bp + 16 + off);
    };

    v8f acc[4];
#pragma unroll
    for (int f = 0; f < 4; ++f)
#pragma unroll
        for (int i = 0; i < 8; ++i) acc[f][i] = 0.f;

    // ---- prologue: fragments for k = 0
    v16h a_c  = ldA(0);
    v16h b_c0 = ldB(0, 0), b_c1 = ldB(1, 0), b_c2 = ldB(2, 0), b_c3 = ldB(3, 0);

    for (int k = 32; k < 512; k += 32) {
        // issue next tile's loads before consuming current fragments
        v16h a_n  = ldA(k);
        v16h b_n0 = ldB(0, k), b_n1 = ldB(1, k), b_n2 = ldB(2, k), b_n3 = ldB(3, k);
        acc[0] = wmma16(a_c, b_c0, acc[0]);
        acc[1] = wmma16(a_c, b_c1, acc[1]);
        acc[2] = wmma16(a_c, b_c2, acc[2]);
        acc[3] = wmma16(a_c, b_c3, acc[3]);
        a_c = a_n; b_c0 = b_n0; b_c1 = b_n1; b_c2 = b_n2; b_c3 = b_n3;
    }
    acc[0] = wmma16(a_c, b_c0, acc[0]);
    acc[1] = wmma16(a_c, b_c1, acc[1]);
    acc[2] = wmma16(a_c, b_c2, acc[2]);
    acc[3] = wmma16(a_c, b_c3, acc[3]);

    // ---- bias now needed: close out the async LDS fill
    asm volatile("s_wait_asynccnt 0x0" ::: "memory");
    __syncthreads();

#pragma unroll
    for (int f = 0; f < 4; ++f) {
        const int n  = n64 + f * 16 + l15;
        const float bb = bias_l[n];
        const int m0 = m16 + roff;                 // first of 8 contiguous rows
        const int b  = m0 >> 10;
        const int s0 = m0 & 1023;
        if (mode == 2) {
            float* outp = (float*)Out + ((size_t)b * 512 + n) * 1024 + s0;
            v8f vv;
#pragma unroll
            for (int i = 0; i < 8; ++i) vv[i] = (acc[f][i] + bb) * scale;
            *(v8f*)outp = vv;
        } else if (mode == 1) {
            const int hh = n >> 6, d = n & 63;
            half_t* outp = (half_t*)Out +
                (((size_t)(b * 8 + hh) * 64 + d) * 1024 + s0);
            v8h hv;
#pragma unroll
            for (int i = 0; i < 8; ++i) hv[i] = (half_t)((acc[f][i] + bb) * scale);
            *(v8h*)outp = hv;
        } else {
            const int hh = n >> 6, d = n & 63;
            half_t* outb = (half_t*)Out + (size_t)(b * 8 + hh) * 1024 * 64 + d;
#pragma unroll
            for (int i = 0; i < 8; ++i)
                outb[(size_t)(s0 + i) * 64] = (half_t)((acc[f][i] + bb) * scale);
        }
    }
}

// ---------------------------------------------------------------------------
// Kernel 5: streaming-softmax attention. Block = 128 thr = 4 waves.
// grid (16 row-blocks of 64, 64 bh). Wave owns 16 query rows; loops over
// 32 kv tiles of 32 keys. Q pre-scaled by 1/8 at projection time.
// Q/K: [bh][s][64] fp16, V: [bh][d][1024] fp16. Output X2: [b*1024+s][512].
// V-fragments (this tile) and K-fragments (next tile) are issued before the
// softmax VALU block so their latency is hidden.
// ---------------------------------------------------------------------------
__global__ __launch_bounds__(128) void attn_k(const half_t* __restrict__ Q,
                                              const half_t* __restrict__ Km,
                                              const half_t* __restrict__ Vt,
                                              half_t* __restrict__ X2) {
    __shared__ half_t plds[4][16 * 40];          // padded P tile per wave
    const int lane = threadIdx.x & 31;
    const int w    = threadIdx.x >> 5;
    const int l15  = lane & 15;
    const int off  = (lane & 16) ? 8 : 0;
    const int roff = (lane & 16) ? 8 : 0;
    const int bh = blockIdx.y;
    const int b = bh >> 3, h = bh & 7;
    const int q0 = blockIdx.x * 64 + w * 16;

    const half_t* Qb = Q  + (size_t)bh * 1024 * 64;
    const half_t* Kb = Km + (size_t)bh * 1024 * 64;
    const half_t* Vb = Vt + (size_t)bh * 64 * 1024;
    half_t* myp = &plds[w][0];

    const half_t* qrow = Qb + (size_t)(q0 + l15) * 64;
    const v16h aq0 = make_frag(qrow + off,      qrow + 16 + off);
    const v16h aq1 = make_frag(qrow + 32 + off, qrow + 48 + off);

    auto ldK = [&](int kb, v16h* kf) {
        const half_t* k0 = Kb + (size_t)(kb + l15) * 64;
        const half_t* k1 = k0 + 16 * 64;
        kf[0] = make_frag(k0 + off,      k0 + 16 + off);
        kf[1] = make_frag(k0 + 32 + off, k0 + 48 + off);
        kf[2] = make_frag(k1 + off,      k1 + 16 + off);
        kf[3] = make_frag(k1 + 32 + off, k1 + 48 + off);
    };

    float mrow[8], lrow[8];
    v8f o[4];
#pragma unroll
    for (int i = 0; i < 8; ++i) { mrow[i] = -1e30f; lrow[i] = 0.f; }
#pragma unroll
    for (int f = 0; f < 4; ++f)
#pragma unroll
        for (int i = 0; i < 8; ++i) o[f][i] = 0.f;

    v16h kf[4];
    ldK(0, kf);                               // prologue: K tile 0

    for (int j = 0; j < 32; ++j) {
        const int kb = j * 32;
        // ---- scores: two 16x16 f32 tiles over 32 keys, K-dim = 64 ----
        v8f c0, c1;
#pragma unroll
        for (int i = 0; i < 8; ++i) { c0[i] = 0.f; c1[i] = 0.f; }
        c0 = wmma16(aq0, kf[0], c0);
        c0 = wmma16(aq1, kf[1], c0);
        c1 = wmma16(aq0, kf[2], c1);
        c1 = wmma16(aq1, kf[3], c1);

        // ---- prefetch: V fragments for this tile, K fragments for next ----
        v16h vf[4];
#pragma unroll
        for (int f = 0; f < 4; ++f) {
            const half_t* vp = Vb + (size_t)(f * 16 + l15) * 1024 + kb;
            vf[f] = make_frag(vp + off, vp + 16 + off);
        }
        ldK(((j + 1) & 31) * 32, kf);         // wraps harmlessly on last iter

        // ---- online softmax (rows in VGPR index, N across 16 lanes) ----
        float corrs[8];
#pragma unroll
        for (int i = 0; i < 8; ++i) {
            float s0 = c0[i];
            float s1 = c1[i];
            float mx = fmaxf(s0, s1);
#pragma unroll
            for (int msk = 8; msk >= 1; msk >>= 1)
                mx = fmaxf(mx, __shfl_xor(mx, msk, 32));
            const float nm = fmaxf(mrow[i], mx);
            const float p0 = __expf(s0 - nm);
            const float p1 = __expf(s1 - nm);
            float rs = p0 + p1;
#pragma unroll
            for (int msk = 8; msk >= 1; msk >>= 1)
                rs += __shfl_xor(rs, msk, 32);
            const float corr = __expf(mrow[i] - nm);
            lrow[i] = lrow[i] * corr + rs;
            mrow[i] = nm;
            corrs[i] = corr;
            myp[(i + roff) * 40 + l15]      = (half_t)p0;
            myp[(i + roff) * 40 + 16 + l15] = (half_t)p1;
        }
#pragma unroll
        for (int f = 0; f < 4; ++f)
#pragma unroll
            for (int i = 0; i < 8; ++i) o[f][i] *= corrs[i];

        asm volatile("s_wait_dscnt 0x0" ::: "memory");  // P store -> P load (wave-local)

        // ---- P (16x32) as A fragment, accumulate P @ V over d=64 ----
        const half_t* pr = myp + l15 * 40;
        const v16h pa = make_frag(pr + off, pr + 16 + off);
        o[0] = wmma16(pa, vf[0], o[0]);
        o[1] = wmma16(pa, vf[1], o[1]);
        o[2] = wmma16(pa, vf[2], o[2]);
        o[3] = wmma16(pa, vf[3], o[3]);
    }

    // ---- finalize and store X2[b*1024+s][h*64+d] ----
#pragma unroll
    for (int f = 0; f < 4; ++f) {
        const int d = f * 16 + l15;
#pragma unroll
        for (int i = 0; i < 8; ++i) {
            const int s = q0 + roff + i;
            const float ov = o[f][i] / lrow[i];
            X2[((size_t)(b * 1024) + s) * 512 + h * 64 + d] = (half_t)ov;
        }
    }
}

// ---------------------------------------------------------------------------
extern "C" void kernel_launch(void* const* d_in, const int* in_sizes, int n_in,
                              void* d_out, int out_size, void* d_ws, size_t ws_size,
                              hipStream_t stream) {
    const float* q    = (const float*)d_in[0];
    const float* gnqw = (const float*)d_in[1];
    const float* gnqb = (const float*)d_in[2];
    const float* gnkw = (const float*)d_in[3];
    const float* gnkb = (const float*)d_in[4];
    const float* gnvw = (const float*)d_in[5];
    const float* gnvb = (const float*)d_in[6];
    const float* Wq   = (const float*)d_in[7];
    const float* bq   = (const float*)d_in[8];
    const float* Wk   = (const float*)d_in[9];
    const float* bk   = (const float*)d_in[10];
    const float* Wv   = (const float*)d_in[11];
    const float* bv   = (const float*)d_in[12];
    const float* Wo   = (const float*)d_in[13];
    const float* bo   = (const float*)d_in[14];

    char* ws = (char*)d_ws;
    size_t o = 0;
    auto take = [&](size_t bytes) {
        size_t cur = o;
        o += (bytes + 255) & ~(size_t)255;
        return cur;
    };
    float*  stats = (float*)(ws + take(256 * 2 * sizeof(float)));
    half_t* Xn    = (half_t*)(ws + take(8192ull * 512 * 2));
    half_t* Btq   = (half_t*)(ws + take(512ull * 512 * 2));
    half_t* Btk   = (half_t*)(ws + take(512ull * 512 * 2));
    half_t* Btv   = (half_t*)(ws + take(512ull * 512 * 2));
    half_t* Bto   = (half_t*)(ws + take(512ull * 512 * 2));
    float*  biasq = (float*)(ws + take(512 * sizeof(float)));
    float*  biask = (float*)(ws + take(512 * sizeof(float)));
    float*  biasv = (float*)(ws + take(512 * sizeof(float)));
    float*  biaso = (float*)(ws + take(512 * sizeof(float)));
    half_t* Qb    = (half_t*)(ws + take(8ull * 8 * 1024 * 64 * 2));
    half_t* Kb    = (half_t*)(ws + take(8ull * 8 * 1024 * 64 * 2));
    half_t* Vt    = (half_t*)(ws + take(8ull * 8 * 64 * 1024 * 2));
    half_t* X2    = Xn;  // Xn is dead after the QKV GEMMs; reuse for attn output

    gn_stats_k<<<256, 256, 0, stream>>>(q, stats);
    fold_w_k<<<512, 256, 0, stream>>>(gnqw, gnqb, Wq, bq, Btq, biasq, 0);
    fold_w_k<<<512, 256, 0, stream>>>(gnkw, gnkb, Wk, bk, Btk, biask, 0);
    fold_w_k<<<512, 256, 0, stream>>>(gnvw, gnvb, Wv, bv, Btv, biasv, 0);
    fold_w_k<<<512, 256, 0, stream>>>(gnqw, gnqb, Wo, bo, Bto, biaso, 1);
    xn_tok_k<<<32, 256, 0, stream>>>(q, stats, Xn);

    dim3 gg(128, 4);
    // 1/sqrt(64) folded into Q projection
    gemm512_k<<<gg, 256, 0, stream>>>(Xn, Btq, biasq, Qb, 0, 0.125f);
    gemm512_k<<<gg, 256, 0, stream>>>(Xn, Btk, biask, Kb, 0, 1.0f);
    gemm512_k<<<gg, 256, 0, stream>>>(Xn, Btv, biasv, Vt, 1, 1.0f);

    attn_k<<<dim3(16, 64), 128, 0, stream>>>(Qb, Kb, Vt, X2);

    gemm512_k<<<gg, 256, 0, stream>>>(X2, Bto, biaso, d_out, 2, 1.0f);
}
